// MoEGate_10660108829478
// MI455X (gfx1250) — compile-verified, matrix operand order checked by reading, not run on hardware
//
#include <hip/hip_runtime.h>
#include <hip/hip_bf16.h>
#include <math.h>

// ---------------- problem constants (from reference) ----------------
#define BATCH      4
#define SEQ        4096
#define HID        2048
#define NEXP       64
#define TOPK       8
#define NTOK       (BATCH * SEQ)        // 16384
#define ALPHA      0.1f

// ---------------- kernel tiling ----------------
#define BLOCK_THREADS 256               // 8 wave32
#define WAVES         8
#define TOK_PER_BLK   (WAVES * 16)      // 128 tokens per block
#define NBLOCKS       (NTOK / TOK_PER_BLK) // 128
#define KCHUNK        128               // K staged per LDS chunk
#define NKCHUNK       (HID / KCHUNK)    // 16
#define WSTRIDE       136               // padded bf16 row stride (272B, 16B aligned)
#define LSTRIDE       68                // padded f32 logits row stride

// output layout (floats, concatenated in reference return order)
#define OFF_IDX    0
#define OFF_W      (NTOK * TOPK)            // 131072
#define OFF_AUX    (2 * NTOK * TOPK)        // 262144
#define OFF_LOADS  (2 * NTOK * TOPK + 1)    // 262145

typedef __attribute__((ext_vector_type(16))) __bf16 v16bf;
typedef __attribute__((ext_vector_type(8)))  __bf16 v8bf;
typedef __attribute__((ext_vector_type(8)))  float  v8f;

union Frag {
    v16bf    v;
    v8bf     h[2];
    uint4    q[2];
};

__global__ __launch_bounds__(BLOCK_THREADS)
void moe_gate_kernel(const float* __restrict__ x,      // [NTOK, HID]
                     const float* __restrict__ w,      // [NEXP, HID]
                     float* __restrict__ out,          // full output buffer
                     float* __restrict__ ws_cnts,      // [BATCH, NEXP]
                     float* __restrict__ ws_sums)      // [BATCH, NEXP]
{
    __shared__ unsigned short wbuf[NEXP * WSTRIDE];          // 17408 B
    __shared__ float logits[TOK_PER_BLK * LSTRIDE];          // 34816 B
    __shared__ float blk_sums[NEXP];
    __shared__ float blk_cnts[NEXP];

    const int tid  = threadIdx.x;
    const int wave = tid >> 5;
    const int lane = tid & 31;
    const int n    = lane & 15;       // column within 16-wide tile
    const int half = lane >> 4;       // K-half selector per WMMA layout

    const long tok_base = (long)blockIdx.x * TOK_PER_BLK;
    const long arow     = tok_base + wave * 16 + n;   // A-matrix row (M = lane%16)

    v8f acc[4];
#pragma unroll
    for (int t = 0; t < 4; ++t)
#pragma unroll
        for (int r = 0; r < 8; ++r) acc[t][r] = 0.0f;

    for (int kc = 0; kc < NKCHUNK; ++kc) {
        __syncthreads();   // previous chunk's B reads complete
        // ---- stage W[:, kc*128 .. +127] into LDS as bf16 ----
        {
            const int l  = tid * 32;                  // 256 threads * 32 = 8192 elems
            const int e  = l >> 7;                    // /128
            const int k0 = l & 127;
            const float* gp = w + (long)e * HID + kc * KCHUNK + k0;
            unsigned short* lp = wbuf + e * WSTRIDE + k0;
#pragma unroll
            for (int j = 0; j < 32; j += 8) {
                v8f f = *(const v8f*)(gp + j);                   // 32B load
                *(v8bf*)(lp + j) = __builtin_convertvector(f, v8bf); // 16B store
            }
        }
        __syncthreads();

        const float* xrow = x + arow * HID + kc * KCHUNK + half * 8;
#pragma unroll
        for (int kk = 0; kk < KCHUNK; kk += 32) {
            // ---- A fragment: 16 tokens x 32 K (f32 -> bf16 via native cvt) ----
            Frag a;
            {
                v8f f0 = *(const v8f*)(xrow + kk);        // K = base + 0..7
                v8f f1 = *(const v8f*)(xrow + kk + 16);   // K = base + 16..23
                a.h[0] = __builtin_convertvector(f0, v8bf);
                a.h[1] = __builtin_convertvector(f1, v8bf);
            }
            // ---- load all 4 B fragments first (partial dscnt waits) ----
            Frag b[4];
#pragma unroll
            for (int t = 0; t < 4; ++t) {
                const unsigned short* bp =
                    wbuf + (t * 16 + n) * WSTRIDE + kk + half * 8;
                b[t].q[0] = *(const uint4*)(bp);
                b[t].q[1] = *(const uint4*)(bp + 16);
            }
#pragma unroll
            for (int t = 0; t < 4; ++t) {
                acc[t] = __builtin_amdgcn_wmma_f32_16x16x32_bf16(
                    false, a.v, false, b[t].v, (short)0, acc[t], false, false);
            }
        }
    }

    // ---- dump logits to LDS: token m = half*8 + r, expert = t*16 + n ----
#pragma unroll
    for (int t = 0; t < 4; ++t)
#pragma unroll
        for (int r = 0; r < 8; ++r) {
            int m = half * 8 + r;
            logits[(wave * 16 + m) * LSTRIDE + t * 16 + n] = acc[t][r];
        }
    if (tid < NEXP) { blk_sums[tid] = 0.0f; blk_cnts[tid] = 0.0f; }
    __syncthreads();

    // ---- per-token softmax + top-8 (one thread per token) ----
    if (tid < TOK_PER_BLK) {
        const float* lrow = &logits[tid * LSTRIDE];
        float s[NEXP];
        float mx = -3.4e38f;
#pragma unroll
        for (int e = 0; e < NEXP; ++e) mx = fmaxf(mx, lrow[e]);
        float sum = 0.0f;
#pragma unroll
        for (int e = 0; e < NEXP; ++e) {
            float v = __expf(lrow[e] - mx);
            s[e] = v;
            sum += v;
        }
        const float inv = 1.0f / sum;
#pragma unroll
        for (int e = 0; e < NEXP; ++e) {
            s[e] *= inv;
            atomicAdd(&blk_sums[e], s[e]);
        }
        // top-8 via masked argmax (bitmask keeps s[] in VGPRs)
        unsigned long long mask = 0ull;
        float tw[TOPK];
        int   ti[TOPK];
        float wsum = 0.0f;
#pragma unroll
        for (int k = 0; k < TOPK; ++k) {
            float bv = -1.0f;
            int   bi = 0;
#pragma unroll
            for (int e = 0; e < NEXP; ++e) {
                bool used = (mask >> e) & 1ull;
                if (!used && s[e] > bv) { bv = s[e]; bi = e; }
            }
            mask |= (1ull << bi);
            tw[k] = bv;
            ti[k] = bi;
            wsum += bv;
            atomicAdd(&blk_cnts[bi], 1.0f);
        }
        const float wn = 1.0f / (wsum + 1e-20f);
        const long tok = tok_base + tid;
#pragma unroll
        for (int k = 0; k < TOPK; ++k) {
            out[OFF_IDX + tok * TOPK + k] = (float)ti[k];
            out[OFF_W   + tok * TOPK + k] = tw[k] * wn;
        }
    }
    __syncthreads();

    // ---- merge block stats into per-(batch, expert) workspace ----
    if (tid < NEXP) {
        const int b = (int)(tok_base / SEQ);
        atomicAdd(&ws_cnts[b * NEXP + tid], blk_cnts[tid]);
        atomicAdd(&ws_sums[b * NEXP + tid], blk_sums[tid]);
    }
}

__global__ void moe_gate_finalize(const float* __restrict__ ws_cnts,
                                  const float* __restrict__ ws_sums,
                                  float* __restrict__ out)
{
    __shared__ float red[NEXP];
    const int e = threadIdx.x;   // 64 threads
    float load = 0.0f, aux = 0.0f;
#pragma unroll
    for (int b = 0; b < BATCH; ++b) {
        float c = ws_cnts[b * NEXP + e];
        load += c;
        float ce = c / ((float)(SEQ * TOPK) / (float)NEXP);   // /512
        float ms = ws_sums[b * NEXP + e] / (float)SEQ;
        aux += ce * ms;
    }
    out[OFF_LOADS + e] = load;
    red[e] = aux;
    __syncthreads();
    for (int off = NEXP / 2; off > 0; off >>= 1) {
        if (e < off) red[e] += red[e + off];
        __syncthreads();
    }
    if (e == 0) out[OFF_AUX] = red[0] * (ALPHA / (float)BATCH);
}

extern "C" void kernel_launch(void* const* d_in, const int* in_sizes, int n_in,
                              void* d_out, int out_size, void* d_ws, size_t ws_size,
                              hipStream_t stream) {
    (void)in_sizes; (void)n_in; (void)out_size; (void)ws_size;
    const float* x = (const float*)d_in[0];   // [4,4096,2048] f32
    const float* w = (const float*)d_in[1];   // [64,2048] f32
    float* out = (float*)d_out;

    float* ws_cnts = (float*)d_ws;                  // [4,64]
    float* ws_sums = ws_cnts + BATCH * NEXP;        // [4,64]

    hipMemsetAsync(d_ws, 0, 2 * BATCH * NEXP * sizeof(float), stream);
    moe_gate_kernel<<<NBLOCKS, BLOCK_THREADS, 0, stream>>>(x, w, out, ws_cnts, ws_sums);
    moe_gate_finalize<<<1, NEXP, 0, stream>>>(ws_cnts, ws_sums, out);
}
